// ShadowAnticipation_13426067768010
// MI455X (gfx1250) — compile-verified, matrix operand order checked by reading, not run on hardware
//
#include <hip/hip_runtime.h>

typedef __attribute__((ext_vector_type(16))) _Float16 v16h;
typedef __attribute__((ext_vector_type(8)))  float    v8f;

#define HF 128
#define WF 128
#define HWF (HF*WF)
#define NB 4
#define NC 512
#define NOFF 6
#define HMAP 1024
#define WMAP 1024

__constant__ float c_offs[NOFF] = {12.f, 20.f, 30.f, 42.f, 55.f, 67.f};

// ---------------------------------------------------------------------------
// Phase 1: fused (bilinear shift  +  8x linear resize) -> f16 evidence tiles.
// resize samples evidence at 8i+3.5 (frac 0.5) => average of bilinear samples
// at integer offsets => separable 3-tap stencil on the shadow map:
//   weights along an axis over {P0, P0+1, P0+2} = {(1-f)/2, 1/2, f/2}
// Zero padding == drop OOB taps. Stored as [b][pix][8] halfs (K padded to 8).
// ---------------------------------------------------------------------------
__global__ __launch_bounds__(256) void evidence_kernel(
    const float* __restrict__ smap,   // (B,1,1024,1024)
    const float* __restrict__ sdir,   // (B,2)
    _Float16*    __restrict__ ev)     // (B, HWF, 8) f16
{
    int idx = blockIdx.x * blockDim.x + threadIdx.x;
    if (idx >= NB * NOFF * HWF) return;
    int pix = idx & (HWF - 1);
    int t   = idx >> 14;              // (b*6 + n)
    int n   = t % NOFF;
    int b   = t / NOFF;
    int w   = pix & (WF - 1);
    int h   = pix >> 7;

    float d   = c_offs[n];
    float dx  =  sdir[2*b + 0] * d;
    float dy  = -sdir[2*b + 1] * d;

    float xs = (float)(8*w + 3) + dx;
    float ys = (float)(8*h + 3) + dy;
    float x0f = floorf(xs), y0f = floorf(ys);
    float fx = xs - x0f,    fy = ys - y0f;
    int   X0 = (int)x0f,    Y0 = (int)y0f;

    float wx0 = 0.5f*(1.f - fx), wx1 = 0.5f, wx2 = 0.5f*fx;
    float wy0 = 0.5f*(1.f - fy), wy1 = 0.5f, wy2 = 0.5f*fy;

    const float* img = smap + (size_t)b * (HMAP*WMAP);
    float acc = 0.f;
#pragma unroll
    for (int j = 0; j < 3; ++j) {
        int Y = Y0 + j;
        if ((unsigned)Y >= (unsigned)HMAP) continue;
        const float* row = img + (size_t)Y * WMAP;
        float wy = (j == 0) ? wy0 : (j == 1) ? wy1 : wy2;
        float r = 0.f;
#pragma unroll
        for (int i = 0; i < 3; ++i) {
            int X = X0 + i;
            if ((unsigned)X >= (unsigned)WMAP) continue;
            float wx = (i == 0) ? wx0 : (i == 1) ? wx1 : wx2;
            r += wx * row[X];
        }
        acc += wy * r;
    }

    _Float16* dst = ev + ((size_t)b * HWF + pix) * 8;
    dst[n] = (_Float16)acc;
    if (n == 0) { dst[6] = (_Float16)0.f; dst[7] = (_Float16)0.f; }
}

// ---------------------------------------------------------------------------
// Phase 2: per wave, D(16pix x 16ch) = A(16x32 f16, K=6 live) x B(32x16 f16)
// via v_wmma_f32_16x16x32_f16; result bounced through LDS to get pixel-major
// coalesced rgb_feat reads / out writes (2x float4 per lane).
//
// A layout (16-bit A 16x32): lane m (<16) holds K=0..7 in halfs 0..7 (one
// 16B load of the padded evidence row), K=16..23 zero; high lanes all zero
// (K=8..15, 24..31 are padding).
// B layout (16-bit B 32x16): lane n (<16) = column c_base+n, half 2v/2v+1 =
// K rows 2v/2v+1 -> halfs 0..5 = w_conv[c][0..5]; high lanes (K=8..15, 24..31) zero.
// ---------------------------------------------------------------------------
__global__ __launch_bounds__(256) void mix_kernel(
    const float*    __restrict__ rgb,    // (B,C,HF,WF)
    const _Float16* __restrict__ ev,     // (B,HWF,8) f16
    const float*    __restrict__ wconv,  // (C,6)
    const float*    __restrict__ bconv,  // (C,)
    const float*    __restrict__ gatep,  // (1,)
    float*          __restrict__ out)    // (B,C,HF,WF)
{
    __shared__ float lds[8][16][17];     // [wave][pixel][channel] (+1 pad)

    int lane = threadIdx.x & 31;
    int wave = threadIdx.x >> 5;

    int bid    = blockIdx.x;             // b(4) * ctile(32) * pgroup(128)
    int pgroup = bid & 127;
    int ctile  = (bid >> 7) & 31;
    int b      = bid >> 12;

    int pix_base = pgroup * 128 + wave * 16;
    int c_base   = ctile * 16;
    int m        = lane & 15;
    bool low     = lane < 16;

    // --- A fragment ---
    v16h a = {};
    if (low) {
        const float4* src =
            (const float4*)(ev + ((size_t)b * HWF + pix_base + m) * 8);
        union { float4 f; _Float16 h[8]; } u;
        u.f = *src;
#pragma unroll
        for (int i = 0; i < 8; ++i) a[i] = u.h[i];
    }

    // --- B fragment ---
    v16h bm = {};
    if (low) {
        const float* ws = wconv + (size_t)(c_base + m) * NOFF;
#pragma unroll
        for (int i = 0; i < NOFF; ++i) bm[i] = (_Float16)ws[i];
    }

    v8f acc = {};
    acc = __builtin_amdgcn_wmma_f32_16x16x32_f16(
        /*neg_a=*/false, a, /*neg_b=*/false, bm,
        /*c_mod=*/(short)0, acc, /*reuse_a=*/false, /*reuse_b=*/false);

    float gate = gatep[0];
    float bias = bconv[c_base + m];      // each lane's column N = lane&15

    int rowoff = low ? 0 : 8;            // D: VGPR v -> pixel row v (+8 high)
#pragma unroll
    for (int v = 0; v < 8; ++v)
        lds[wave][v + rowoff][m] = gate * (acc[v] + bias);

    __syncthreads();

    // Re-read pixel-major: lane -> channel ch = lane/2, pixels p0..p0+7
    int ch = lane >> 1;
    int p0 = (lane & 1) * 8;
    size_t gidx = ((size_t)(b * NC + c_base + ch)) * HWF + pix_base + p0;

    float4 r0 = *(const float4*)(rgb + gidx);
    float4 r1 = *(const float4*)(rgb + gidx + 4);
    float4 o0, o1;
    o0.x = r0.x + lds[wave][p0 + 0][ch];
    o0.y = r0.y + lds[wave][p0 + 1][ch];
    o0.z = r0.z + lds[wave][p0 + 2][ch];
    o0.w = r0.w + lds[wave][p0 + 3][ch];
    o1.x = r1.x + lds[wave][p0 + 4][ch];
    o1.y = r1.y + lds[wave][p0 + 5][ch];
    o1.z = r1.z + lds[wave][p0 + 6][ch];
    o1.w = r1.w + lds[wave][p0 + 7][ch];
    *(float4*)(out + gidx)     = o0;
    *(float4*)(out + gidx + 4) = o1;
}

extern "C" void kernel_launch(void* const* d_in, const int* in_sizes, int n_in,
                              void* d_out, int out_size, void* d_ws, size_t ws_size,
                              hipStream_t stream) {
    const float* rgb   = (const float*)d_in[0];
    const float* smap  = (const float*)d_in[1];
    const float* sdir  = (const float*)d_in[2];
    const float* wconv = (const float*)d_in[3];
    const float* bconv = (const float*)d_in[4];
    const float* gate  = (const float*)d_in[5];
    float* out = (float*)d_out;

    _Float16* ev = (_Float16*)d_ws;      // NB*HWF*8 halfs = 1 MiB

    int n1 = NB * NOFF * HWF;            // 393216 threads
    evidence_kernel<<<(n1 + 255) / 256, 256, 0, stream>>>(smap, sdir, ev);

    // 4 b * 32 ctiles * 128 pgroups = 16384 blocks, 8 waves each = 1 tile/wave
    mix_kernel<<<NB * 32 * 128, 256, 0, stream>>>(rgb, ev, wconv, bconv, gate, out);
}
